// AutoIntLayer_26456998543817
// MI455X (gfx1250) — matrix operand order repaired and checked by reading, not compiled, not run on hardware
//
#include <hip/hip_runtime.h>

// Fused AutoInt (2 layers, shared weights) for MI455X / gfx1250.
// Two wave32s cooperate on one batch element out of LDS (wave w owns head w,
// and half the projections / rows). 8 waves per WG at 263 KB LDS -> 1 WG/WGP
// but 2 waves/SIMD for latency hiding. HBM traffic is x in + out only
// (2 x 167.8 MB -> ~14 us roofline @ 23.3 TB/s).
// GEMMs: v_wmma_f32_16x16x32_f16. Softmax/LayerNorm are lane-per-row.

typedef __attribute__((ext_vector_type(16))) _Float16 v16h;
typedef __attribute__((ext_vector_type(8)))  _Float16 v8h;
typedef __attribute__((ext_vector_type(4)))  _Float16 v4h;
typedef __attribute__((ext_vector_type(8)))  float    v8f;

#define NBATCH   16384
#define FSEQ     40
#define FPAD     48
#define EDIM     64
#define NHEAD    2
#define DHEAD    32
#define NLAYER   2
#define LN_EPS_F 1e-3f
#define BPW      4                       // batches per workgroup (2 waves each)
#define NTHREADS (BPW * 64)

// per-batch LDS layout (halves unless noted)
#define CUR_OFF  0                       // 48x64 f16 activations
#define QB_OFF   (CUR_OFF + FPAD*EDIM)   // 48x64 f16
#define KB_OFF   (QB_OFF  + FPAD*EDIM)   // 48x64 f16
#define VB_OFF   (KB_OFF  + FPAD*EDIM)   // 64x64 f16 (rows>=40 zero: K-pad)
#define RB_OFF   (VB_OFF  + EDIM*EDIM)   // 48x64 f16
#define SB_OFF   (RB_OFF  + FPAD*EDIM)   // 2 x 48x64 f16 (per-head scores/probs)
#define ZERO_H   (SB_OFF  + 2*FPAD*EDIM) // halves to zero-init
#define OB_OFF   ZERO_H                  // 48x64 f32 attention output (bytes *2)
#define BATCH_BYTES (ZERO_H*2 + FPAD*EDIM*4)                     // 57344
#define HDR_BYTES   (4*EDIM*EDIM*2 + 4*EDIM*2 + 2*EDIM*4)        // 33792
#define SMEM_BYTES  (HDR_BYTES + BPW*BATCH_BYTES)                // 263168 <= 320K

__device__ __forceinline__ void lds_fence() {
  asm volatile("s_wait_dscnt 0" ::: "memory");
}

__device__ __forceinline__ v8f wmma_f16(v16h a, v16h b, v8f c) {
  return __builtin_amdgcn_wmma_f32_16x16x32_f16(false, a, false, b, (short)0, c,
                                                false, false);
}

// A-fragment (16x32 f16), row-major LDS, ld=EDIM.
// lane&15 = M row, lane>>4 = K-half; half h -> K = (h&7)+(h>=8?16:0)+8*khalf
__device__ __forceinline__ v16h lda(const _Float16* buf, int row0, int col0, int lane) {
  const int m  = row0 + (lane & 15);
  const int kh = (lane >> 4) << 3;
  v16h a;
#pragma unroll
  for (int h = 0; h < 16; ++h) {
    const int k = (h & 7) + ((h >> 3) << 4) + kh;
    a[h] = buf[m * EDIM + col0 + k];
  }
  return a;
}

// B-fragment (32x16 f16): B[k][n] = buf[(krow0+k)*EDIM + col0 + n]; lane&15 = n, K order as lda
__device__ __forceinline__ v16h ldb(const _Float16* buf, int krow0, int col0, int lane) {
  const int n  = col0 + (lane & 15);
  const int kh = (lane >> 4) << 3;
  v16h b;
#pragma unroll
  for (int h = 0; h < 16; ++h) {
    const int k = krow0 + (h & 7) + ((h >> 3) << 4) + kh;
    b[h] = buf[k * EDIM + n];
  }
  return b;
}

// B-fragment of a transposed tile: B[k][n] = buf[(n0+n)*EDIM + col0 + k], lane&15 = n, K order as lda
__device__ __forceinline__ v16h ldbt(const _Float16* buf, int n0, int col0, int lane) {
  const int n  = n0 + (lane & 15);
  const int kh = (lane >> 4) << 3;
  v16h b;
#pragma unroll
  for (int h = 0; h < 16; ++h) {
    const int k = (h & 7) + ((h >> 3) << 4) + kh;
    b[h] = buf[n * EDIM + col0 + k];
  }
  return b;
}

__global__ void __launch_bounds__(NTHREADS) autoint_fused(
    const float* __restrict__ x,
    const float* __restrict__ Wq, const float* __restrict__ bq,
    const float* __restrict__ Wk, const float* __restrict__ bk,
    const float* __restrict__ Wv, const float* __restrict__ bv,
    const float* __restrict__ Wr, const float* __restrict__ br,
    const float* __restrict__ gamma, const float* __restrict__ beta,
    float* __restrict__ out)
{
  extern __shared__ char smem[];
  _Float16* Wsh  = (_Float16*)smem;                // 4 x 64x64 f16 weights
  _Float16* Bsh  = Wsh + 4*EDIM*EDIM;              // 4 x 64 f16 biases
  float*    Gsh  = (float*)(Bsh + 4*EDIM);         // gamma (f32)
  float*    Besh = Gsh + EDIM;                     // beta  (f32)

  const int tid   = threadIdx.x;
  const int wave  = tid >> 5;
  const int lane  = tid & 31;
  const int wv    = wave & 1;      // 0/1 within the pair owning one batch
  const int pair  = wave >> 1;     // which batch within the WG
  const int lane2 = lane + wv * 32;  // 0..63 across the pair

  char* bb = smem + HDR_BYTES + (size_t)pair * BATCH_BYTES;
  _Float16* cur = (_Float16*)bb + CUR_OFF;
  _Float16* qb  = (_Float16*)bb + QB_OFF;
  _Float16* kb  = (_Float16*)bb + KB_OFF;
  _Float16* vb  = (_Float16*)bb + VB_OFF;
  _Float16* rb  = (_Float16*)bb + RB_OFF;
  _Float16* sbm = (_Float16*)bb + SB_OFF + wv * (FPAD*EDIM);  // this wave's head
  float*    ob  = (float*)(bb + OB_OFF * 2);

  // ---- shared weights: block-cooperative load, fp32 -> f16 ----
  for (int i = tid; i < 4*EDIM*EDIM; i += NTHREADS) {
    const int p = i >> 12;
    const float* w = (p == 0) ? Wq : (p == 1) ? Wk : (p == 2) ? Wv : Wr;
    Wsh[i] = (_Float16)w[i & 4095];
  }
  for (int i = tid; i < 4*EDIM; i += NTHREADS) {
    const int p = i >> 6;
    const float* b = (p == 0) ? bq : (p == 1) ? bk : (p == 2) ? bv : br;
    Bsh[i] = (_Float16)b[i & 63];
  }
  for (int i = tid; i < EDIM; i += NTHREADS) { Gsh[i] = gamma[i]; Besh[i] = beta[i]; }

  const int batch = blockIdx.x * BPW + pair;
  const float* xb = x + (size_t)batch * (FSEQ * EDIM);

  // prefetch this pair's 10 KB input block while we zero LDS
  __builtin_prefetch(xb + lane2 * 32, 0, 1);
  __builtin_prefetch(xb + (lane2 + 16) * 32, 0, 1);

  // zero the f16 scratch of this batch (pair-split)
  {
    unsigned* z = (unsigned*)cur;
    for (int i = lane2; i < (ZERO_H >> 1); i += 64) z[i] = 0u;
  }
  __syncthreads();

  // ---- stage x -> cur (f16) with float4 loads (pair-split) ----
  for (int i = lane2 * 4; i < FSEQ * EDIM; i += 256) {
    const float4 x4 = *(const float4*)(xb + i);
    v4h h;
    h[0] = (_Float16)x4.x; h[1] = (_Float16)x4.y;
    h[2] = (_Float16)x4.z; h[3] = (_Float16)x4.w;
    *(v4h*)(cur + i) = h;
  }
  __syncthreads();

  const float scale = 0.17677669529663687f;   // 1/sqrt(Dh)
  const int   hw8   = (lane >> 4) << 3;       // C-fragment row offset
  const int   cl    = lane & 15;              // C-fragment column
  const int   c0    = wv * DHEAD;             // this wave's head column base

  for (int layer = 0; layer < NLAYER; ++layer) {
    // ======== 1) projections: wave0 -> q,k   wave1 -> v,r ========
#pragma unroll
    for (int pp = 0; pp < 2; ++pp) {
      const int p = wv * 2 + pp;
      const _Float16* W    = Wsh + p * EDIM * EDIM;
      const _Float16* bvec = Bsh + p * EDIM;
      _Float16* dst = (p == 0) ? qb : (p == 1) ? kb : (p == 2) ? vb : rb;
      for (int mt = 0; mt < 3; ++mt) {
        const v16h a0 = lda(cur, mt * 16, 0,  lane);
        const v16h a1 = lda(cur, mt * 16, 32, lane);
#pragma unroll
        for (int nt = 0; nt < 4; ++nt) {
          v8f acc = {};
          acc = wmma_f16(a0, ldb(W, 0,  nt * 16, lane), acc);
          acc = wmma_f16(a1, ldb(W, 32, nt * 16, lane), acc);
          const int n  = nt * 16 + cl;
          const float bn = (float)bvec[n];
#pragma unroll
          for (int r = 0; r < 8; ++r) {
            const int m = mt * 16 + r + hw8;
            float v = acc[r] + bn;
            v = v > 0.f ? v : 0.f;
            dst[m * EDIM + n] = (m < FSEQ) ? (_Float16)v : (_Float16)0.f;
          }
        }
      }
    }
    __syncthreads();   // q,k,v,r complete across the pair

    // ======== 2) attention: this wave owns head wv ========
    // raw scores Qh.Kh^T -> sbm (f16, cols 48..63 stay zero)
    for (int mt = 0; mt < 3; ++mt) {
      const v16h aq = lda(qb, mt * 16, c0, lane);
#pragma unroll
      for (int nt = 0; nt < 3; ++nt) {
        v8f sc = {};
        sc = wmma_f16(aq, ldbt(kb, nt * 16, c0, lane), sc);
        const int n = nt * 16 + cl;
#pragma unroll
        for (int r = 0; r < 8; ++r)
          sbm[(mt * 16 + r + hw8) * EDIM + n] = (_Float16)sc[r];
      }
    }
    lds_fence();

    // lane-per-row masked softmax, in place in sbm (f16 -> f16)
#pragma unroll
    for (int pass = 0; pass < 2; ++pass) {
      const int row = lane + 32 * pass;
      if (row < FSEQ) {
        float sv[FSEQ];
        float mx = -3.0e38f;
#pragma unroll
        for (int j8 = 0; j8 < FSEQ / 8; ++j8) {
          const v8h s8 = *(const v8h*)(sbm + row * EDIM + j8 * 8);
#pragma unroll
          for (int e = 0; e < 8; ++e) {
            const float v = (float)s8[e];
            sv[j8 * 8 + e] = v;
            mx = fmaxf(mx, v);
          }
        }
        float sum = 0.f;
#pragma unroll
        for (int j = 0; j < FSEQ; ++j) {
          const float e = __expf((sv[j] - mx) * scale);
          sv[j] = e;
          sum += e;
        }
        const float inv = 1.0f / sum;
#pragma unroll
        for (int j8 = 0; j8 < 8; ++j8) {
          v8h hv;
#pragma unroll
          for (int e = 0; e < 8; ++e) {
            const int c = j8 * 8 + e;
            hv[e] = (c < FSEQ) ? (_Float16)(sv[c] * inv) : (_Float16)0.f;
          }
          *(v8h*)(sbm + row * EDIM + j8 * 8) = hv;
        }
      }
    }
    lds_fence();

    // o_h = softmax @ Vh (K padded to 64; vb pad rows are zero)
    v8f of[6];
#pragma unroll
    for (int mt = 0; mt < 3; ++mt) {
      const v16h a0 = lda(sbm, mt * 16, 0,  lane);
      const v16h a1 = lda(sbm, mt * 16, 32, lane);
#pragma unroll
      for (int dt = 0; dt < 2; ++dt) {
        v8f acc = {};
        acc = wmma_f16(a0, ldb(vb, 0,  c0 + dt * 16, lane), acc);
        acc = wmma_f16(a1, ldb(vb, 32, c0 + dt * 16, lane), acc);
        of[mt * 2 + dt] = acc;
      }
    }
    // spill this head's o fragments into its 32 columns of ob
#pragma unroll
    for (int idx = 0; idx < 6; ++idx) {
      const int mt = idx >> 1, dt = idx & 1;
      const int n = c0 + dt * 16 + cl;
#pragma unroll
      for (int r = 0; r < 8; ++r)
        ob[(mt * 16 + r + hw8) * EDIM + n] = of[idx][r];
    }
    __syncthreads();   // merged heads visible to the pair

    // ======== 3) lane-per-row relu(o + res) -> LayerNorm ========
    const bool last = (layer == NLAYER - 1);
    {
      const int row = 2 * lane + wv;        // interleaved rows across the pair
      if (row < FSEQ) {
        float u[EDIM];
        float ssum = 0.f;
#pragma unroll
        for (int j8 = 0; j8 < 8; ++j8) {
          const v8h r8 = *(const v8h*)(rb + row * EDIM + j8 * 8);
#pragma unroll
          for (int e = 0; e < 8; ++e) {
            float v = ob[row * EDIM + j8 * 8 + e] + (float)r8[e];
            v = v > 0.f ? v : 0.f;
            u[j8 * 8 + e] = v;
            ssum += v;
          }
        }
        const float mu = ssum * (1.0f / EDIM);
        float vs = 0.f;
#pragma unroll
        for (int j = 0; j < EDIM; ++j) {
          const float d = u[j] - mu;
          vs += d * d;
        }
        const float rstd = rsqrtf(vs * (1.0f / EDIM) + LN_EPS_F);
        if (last) {
#pragma unroll
          for (int j = 0; j < EDIM / 4; ++j) {
            const float4 g4 = ((const float4*)Gsh)[j];
            const float4 b4 = ((const float4*)Besh)[j];
            float4 y;
            y.x = (u[4*j+0] - mu) * rstd * g4.x + b4.x;
            y.y = (u[4*j+1] - mu) * rstd * g4.y + b4.y;
            y.z = (u[4*j+2] - mu) * rstd * g4.z + b4.z;
            y.w = (u[4*j+3] - mu) * rstd * g4.w + b4.w;
            ((float4*)(ob + row * EDIM))[j] = y;
          }
        } else {
#pragma unroll
          for (int j8 = 0; j8 < 8; ++j8) {
            v8h hv;
#pragma unroll
            for (int e = 0; e < 8; ++e) {
              const int c = j8 * 8 + e;
              hv[e] = (_Float16)((u[c] - mu) * rstd * Gsh[c] + Besh[c]);
            }
            *(v8h*)(cur + row * EDIM + j8 * 8) = hv;
          }
        }
      }
    }
    __syncthreads();   // cur (or normalized ob) complete for the pair

    if (last) {
      // coalesced float4 copy of normalized rows: ob -> HBM (pair-split)
      float* outb = out + (size_t)batch * (FSEQ * EDIM);
      for (int i = lane2 * 4; i < FSEQ * EDIM; i += 256)
        *(float4*)(outb + i) = *(const float4*)(ob + i);
    }
  }
}

extern "C" void kernel_launch(void* const* d_in, const int* in_sizes, int n_in,
                              void* d_out, int out_size, void* d_ws, size_t ws_size,
                              hipStream_t stream) {
  (void)in_sizes; (void)n_in; (void)out_size; (void)d_ws; (void)ws_size;
  const float* x     = (const float*)d_in[0];
  const float* Wq    = (const float*)d_in[1];
  const float* bq    = (const float*)d_in[2];
  const float* Wk    = (const float*)d_in[3];
  const float* bk    = (const float*)d_in[4];
  const float* Wv    = (const float*)d_in[5];
  const float* bv    = (const float*)d_in[6];
  const float* Wr    = (const float*)d_in[7];
  const float* br    = (const float*)d_in[8];
  const float* gamma = (const float*)d_in[9];
  const float* beta  = (const float*)d_in[10];
  float* outp = (float*)d_out;

  dim3 grid(NBATCH / BPW);
  dim3 block(NTHREADS);
  autoint_fused<<<grid, block, SMEM_BYTES, stream>>>(
      x, Wq, bq, Wk, bk, Wv, bv, Wr, br, gamma, beta, outp);
}
